// Channel_Seq_Big_Attention_89232240541886
// MI455X (gfx1250) — matrix operand order repaired and checked by reading, not compiled, not run on hardware
//
#include <hip/hip_runtime.h>

// Channel/Seq attention for MI455X (gfx1250, wave32, WMMA f32<=f16 16x16x32).
// Pipeline: cvt(x,W->f16) -> proj GEMM (Q,K,Vt) -> 512x per-(b,h,z) attention
//           -> out-proj GEMM with bias. All matrix math via v_wmma_f32_16x16x32_f16.
// GEMM mainloops are double-buffered in LDS (one barrier/K-iter, global loads
// for tile k+1 issued before the WMMA burst on tile k).

typedef _Float16 half_t;
typedef __attribute__((ext_vector_type(16))) _Float16 v16h;
typedef __attribute__((ext_vector_type(8)))  _Float16 v8h;
typedef __attribute__((ext_vector_type(8)))  float    v8f;

#define WMMA(a, b, c) __builtin_amdgcn_wmma_f32_16x16x32_f16(false, (a), false, (b), (short)0, (c), false, false)

// ---- Fragment loaders (contiguous K order k = 16*half + s, same for A and B) -------------
// A 16x32 f16: lane(<16)=row r, K = half*16..+15
static __device__ inline v16h load_frag_a(const half_t* base, int ld, int row0, int k0) {
  const int lane = threadIdx.x & 31;
  const int hf = lane >> 4, r = lane & 15;
  const half_t* p = base + (size_t)(row0 + r) * ld + k0 + hf * 16;
  v8h lo = *(const v8h*)(p);
  v8h hi = *(const v8h*)(p + 8);
  return __builtin_shufflevector(lo, hi, 0,1,2,3,4,5,6,7,8,9,10,11,12,13,14,15);
}
// B 32x16 f16 from B^T storage Bt[n][k]: lane col n=lane%16, K = (lane/16)*16..+15
static __device__ inline v16h load_frag_b(const half_t* base, int ld, int n0, int k0) {
  const int lane = threadIdx.x & 31;
  const half_t* p = base + (size_t)(n0 + (lane & 15)) * ld + k0 + (lane >> 4) * 16;
  return *(const v16h*)p;
}

// ---- f32 -> f16 conversion kernels -----------------------------------------
__global__ void cvt_x_kernel(const float* __restrict__ x, half_t* __restrict__ xh, int n) {
  int i = blockIdx.x * blockDim.x + threadIdx.x;
  if (i < n) xh[i] = (half_t)x[i];
}
// Wt[n][k] (n<1536, k<512): n<512 -> wq[k][n]; else wkv[k][n-512]
__global__ void cvt_wqkv_kernel(const float* __restrict__ wq, const float* __restrict__ wkv,
                                half_t* __restrict__ wt) {
  int i = blockIdx.x * blockDim.x + threadIdx.x;
  if (i >= 1536 * 512) return;
  int n = i >> 9, k = i & 511;
  float v = (n < 512) ? wq[(size_t)k * 512 + n] : wkv[(size_t)k * 1024 + (n - 512)];
  wt[i] = (half_t)v;
}
// Wot[n][k] = w_out[k][n], 4096x4096
__global__ void cvt_wout_kernel(const float* __restrict__ w, half_t* __restrict__ wt) {
  int i = blockIdx.x * blockDim.x + threadIdx.x;
  if (i >= 4096 * 4096) return;
  int n = i >> 12, k = i & 4095;
  wt[i] = (half_t)w[(size_t)k * 4096 + n];
}

// ---- Shared double-buffered GEMM mainloop -----------------------------------
// Block tile 128x128, 8 waves in 4x2 grid (32 rows x 64 cols per wave).
// Ag = A base + row0*lda (row-major, K contiguous); Bg = B^T base + col0*ldb.
template <int KTOT>
static __device__ inline void gemm_mainloop(const half_t* __restrict__ Ag,
                                            const half_t* __restrict__ Bg,
                                            int lda, int ldb,
                                            half_t* As, half_t* Bs,  // each 2*128*32
                                            v8f acc[2][4]) {
  const int tid = threadIdx.x;
  const int wave = tid >> 5;
  const int waveR = wave >> 1, waveC = wave & 1;
  const int r = tid >> 1, seg = (tid & 1) * 16;
  const half_t* ap = Ag + (size_t)r * lda + seg;
  const half_t* bp = Bg + (size_t)r * ldb + seg;

  // stage tile k=0
  v8h ra0 = *(const v8h*)(ap);
  v8h ra1 = *(const v8h*)(ap + 8);
  v8h rb0 = *(const v8h*)(bp);
  v8h rb1 = *(const v8h*)(bp + 8);
  *(v8h*)&As[r * 32 + seg] = ra0;
  *(v8h*)&As[r * 32 + seg + 8] = ra1;
  *(v8h*)&Bs[r * 32 + seg] = rb0;
  *(v8h*)&Bs[r * 32 + seg + 8] = rb1;
  __syncthreads();

  for (int kt = 0; kt < KTOT; kt += 32) {
    const int buf = (kt >> 5) & 1;
    half_t* Ab = As + buf * (128 * 32);
    half_t* Bb = Bs + buf * (128 * 32);
    const bool notlast = (kt + 32 < KTOT);
    if (notlast) {  // issue next tile's global loads before the WMMA burst
      ra0 = *(const v8h*)(ap + kt + 32);
      ra1 = *(const v8h*)(ap + kt + 40);
      rb0 = *(const v8h*)(bp + kt + 32);
      rb1 = *(const v8h*)(bp + kt + 40);
    }
    v16h af0 = load_frag_a(Ab, 32, waveR * 32, 0);
    v16h af1 = load_frag_a(Ab, 32, waveR * 32 + 16, 0);
    v16h bf[4];
    #pragma unroll
    for (int ct = 0; ct < 4; ++ct) bf[ct] = load_frag_b(Bb, 32, waveC * 64 + ct * 16, 0);
    #pragma unroll
    for (int ct = 0; ct < 4; ++ct) {
      acc[0][ct] = WMMA(af0, bf[ct], acc[0][ct]);
      acc[1][ct] = WMMA(af1, bf[ct], acc[1][ct]);
    }
    if (notlast) {
      half_t* An = As + (buf ^ 1) * (128 * 32);
      half_t* Bn = Bs + (buf ^ 1) * (128 * 32);
      *(v8h*)&An[r * 32 + seg] = ra0;
      *(v8h*)&An[r * 32 + seg + 8] = ra1;
      *(v8h*)&Bn[r * 32 + seg] = rb0;
      *(v8h*)&Bn[r * 32 + seg + 8] = rb1;
      __syncthreads();  // single barrier per K-iteration
    }
  }
}

// ---- Projection GEMM: [8192x512] x [512x1536] -> scatter to Q, K, Vt (f16) --
__global__ __launch_bounds__(256)
void proj_kernel(const half_t* __restrict__ Xh, const half_t* __restrict__ Wt,
                 half_t* __restrict__ Qb, half_t* __restrict__ Kb, half_t* __restrict__ Vt) {
  __shared__ __align__(16) half_t As[2 * 128 * 32];
  __shared__ __align__(16) half_t Bs[2 * 128 * 32];
  const int row0 = blockIdx.x * 128;
  const int col0 = blockIdx.y * 128;
  const int tid = threadIdx.x;
  const int wave = tid >> 5;
  const int waveR = wave >> 1, waveC = wave & 1;

  v8f acc[2][4];
  #pragma unroll
  for (int a = 0; a < 2; ++a)
    #pragma unroll
    for (int b = 0; b < 4; ++b) acc[a][b] = {};

  gemm_mainloop<512>(Xh + (size_t)row0 * 512, Wt + (size_t)col0 * 512, 512, 512, As, Bs, acc);

  const int lane = tid & 31;
  const int hf = lane >> 4, cl = lane & 15;
  #pragma unroll
  for (int rt = 0; rt < 2; ++rt)
    #pragma unroll
    for (int ct = 0; ct < 4; ++ct)
      #pragma unroll
      for (int r = 0; r < 8; ++r) {
        int R = row0 + waveR * 32 + rt * 16 + r + 8 * hf;  // token = (b*128+i)*8+m
        int C = col0 + waveC * 64 + ct * 16 + cl;          // 0..1535
        float v = acc[rt][ct][r];
        int b = R >> 10, rem = R & 1023, i = rem >> 3, m = rem & 7;
        if (C < 512) {  // Q (scale folded in): [b][h][m][i][dh]
          int h = C >> 6, dh = C & 63;
          Qb[(((size_t)(b * 8 + h) * 8 + m) * 128 + i) * 64 + dh] = (half_t)(v * 0.125f);
        } else if (C < 1024) {  // K: [b][h][z=m][j=i][dh]
          int c2 = C - 512; int h = c2 >> 6, dh = c2 & 63;
          Kb[(((size_t)(b * 8 + h) * 8 + m) * 128 + i) * 64 + dh] = (half_t)v;
        } else {  // V transposed: [b][h][m][dh][j=i]
          int c2 = C - 1024; int h = c2 >> 6, dh = c2 & 63;
          Vt[(((size_t)(b * 8 + h) * 8 + m) * 64 + dh) * 128 + i] = (half_t)v;
        }
      }
}

// ---- Attention: one block per (b,h,z); out_z = sum_m softmax(Q_m K_z^T) V_m --
__global__ __launch_bounds__(256)
void attn_kernel(const half_t* __restrict__ Qb, const half_t* __restrict__ Kb,
                 const half_t* __restrict__ Vt, half_t* __restrict__ AO) {
  __shared__ __align__(16) half_t Klds[128 * 64];      // K_z, Bt layout [j][dh]
  __shared__ __align__(16) half_t Plds[8 * 16 * 128];  // per-wave P strip
  const int idx = blockIdx.x;
  const int b = idx >> 6, h = (idx >> 3) & 7, z = idx & 7;
  const int tid = threadIdx.x, wave = tid >> 5, lane = tid & 31;
  const int hf = lane >> 4, cl = lane & 15;

  {
    const half_t* Kg = Kb + ((size_t)(b * 8 + h) * 8 + z) * 128 * 64;
    #pragma unroll
    for (int v = 0; v < 4; ++v)
      *(v8h*)&Klds[tid * 32 + v * 8] = *(const v8h*)&Kg[tid * 32 + v * 8];
  }
  __syncthreads();

  half_t* Pw = &Plds[wave * (16 * 128)];
  v8f acco[4];
  #pragma unroll
  for (int t = 0; t < 4; ++t) acco[t] = {};

  for (int m = 0; m < 8; ++m) {
    const half_t* Qg = Qb + ((size_t)(b * 8 + h) * 8 + m) * 128 * 64;
    const half_t* Vg = Vt + ((size_t)(b * 8 + h) * 8 + m) * 64 * 128;

    // S(16x128) = Q_m rows of this wave @ K_z^T  (K-dim 64 = 2 wmma steps)
    v8f s[8];
    #pragma unroll
    for (int t = 0; t < 8; ++t) s[t] = {};
    v16h aq0 = load_frag_a(Qg, 64, wave * 16, 0);
    v16h aq1 = load_frag_a(Qg, 64, wave * 16, 32);
    #pragma unroll
    for (int t = 0; t < 8; ++t) {
      v16h bk0 = load_frag_b(Klds, 64, t * 16, 0);
      v16h bk1 = load_frag_b(Klds, 64, t * 16, 32);
      s[t] = WMMA(aq0, bk0, s[t]);
      s[t] = WMMA(aq1, bk1, s[t]);
    }

    // softmax over j (row = vgpr r + 8*half spans 16 lanes x 8 tiles)
    #pragma unroll
    for (int r = 0; r < 8; ++r) {
      float mx = -3.4e38f;
      #pragma unroll
      for (int t = 0; t < 8; ++t) mx = fmaxf(mx, s[t][r]);
      #pragma unroll
      for (int off = 1; off < 16; off <<= 1) mx = fmaxf(mx, __shfl_xor(mx, off, 32));
      float sum = 0.0f;
      #pragma unroll
      for (int t = 0; t < 8; ++t) { float p = __expf(s[t][r] - mx); s[t][r] = p; sum += p; }
      #pragma unroll
      for (int off = 1; off < 16; off <<= 1) sum += __shfl_xor(sum, off, 32);
      float inv = 1.0f / sum;
      #pragma unroll
      for (int t = 0; t < 8; ++t)
        Pw[(r + 8 * hf) * 128 + t * 16 + cl] = (half_t)(s[t][r] * inv);
    }

    // O += P(16x128) @ V_m(128x64); V already B^T-laid-out
    #pragma unroll
    for (int ks = 0; ks < 4; ++ks) {
      v16h ap = load_frag_a(Pw, 128, 0, ks * 32);
      #pragma unroll
      for (int t = 0; t < 4; ++t) {
        v16h bv = load_frag_b(Vg, 128, t * 16, ks * 32);
        acco[t] = WMMA(ap, bv, acco[t]);
      }
    }
  }

  // AO[b*128+i][z*512 + h*64 + dh]
  #pragma unroll
  for (int t = 0; t < 4; ++t)
    #pragma unroll
    for (int r = 0; r < 8; ++r) {
      int i = wave * 16 + r + 8 * hf;
      int dh = t * 16 + cl;
      AO[(size_t)(b * 128 + i) * 4096 + z * 512 + h * 64 + dh] = (half_t)acco[t][r];
    }
}

// ---- Output projection: [1024x4096] x [4096x4096] + bias -> f32 out ---------
__global__ __launch_bounds__(256)
void outproj_kernel(const half_t* __restrict__ AO, const half_t* __restrict__ Wot,
                    const float* __restrict__ bias, float* __restrict__ Y) {
  __shared__ __align__(16) half_t As[2 * 128 * 32];
  __shared__ __align__(16) half_t Bs[2 * 128 * 32];
  const int row0 = blockIdx.x * 128;
  const int col0 = blockIdx.y * 128;
  const int tid = threadIdx.x;
  const int wave = tid >> 5;
  const int waveR = wave >> 1, waveC = wave & 1;

  v8f acc[2][4];
  #pragma unroll
  for (int a = 0; a < 2; ++a)
    #pragma unroll
    for (int b = 0; b < 4; ++b) acc[a][b] = {};

  gemm_mainloop<4096>(AO + (size_t)row0 * 4096, Wot + (size_t)col0 * 4096, 4096, 4096, As, Bs, acc);

  const int lane = tid & 31;
  const int hf = lane >> 4, cl = lane & 15;
  #pragma unroll
  for (int rt = 0; rt < 2; ++rt)
    #pragma unroll
    for (int ct = 0; ct < 4; ++ct)
      #pragma unroll
      for (int r = 0; r < 8; ++r) {
        int R = row0 + waveR * 32 + rt * 16 + r + 8 * hf;
        int C = col0 + waveC * 64 + ct * 16 + cl;
        Y[(size_t)R * 4096 + C] = acc[rt][ct][r] + bias[C];
      }
}

// ---- Host launch -------------------------------------------------------------
extern "C" void kernel_launch(void* const* d_in, const int* in_sizes, int n_in,
                              void* d_out, int out_size, void* d_ws, size_t ws_size,
                              hipStream_t stream) {
  const float* x     = (const float*)d_in[0];
  const float* w_q   = (const float*)d_in[1];
  const float* w_kv  = (const float*)d_in[2];
  const float* w_out = (const float*)d_in[3];
  const float* b_out = (const float*)d_in[4];
  float* y = (float*)d_out;

  char* ws = (char*)d_ws;
  // workspace layout (bytes)
  half_t* Xh  = (half_t*)(ws);             //  8192*512  f16 = 8 MB
  half_t* Wt  = (half_t*)(ws + 8388608);   //  1536*512  f16 = 1.5 MB
  half_t* Wot = (half_t*)(ws + 9961472);   //  4096*4096 f16 = 32 MB
  half_t* Qb  = (half_t*)(ws + 43515904);  //  8 MB
  half_t* Kb  = (half_t*)(ws + 51904512);  //  8 MB
  half_t* Vt  = (half_t*)(ws + 60293120);  //  8 MB
  half_t* AO  = (half_t*)(ws + 68681728);  //  8 MB   (total ~73.5 MB)

  const int nx = 8192 * 512;
  cvt_x_kernel<<<(nx + 255) / 256, 256, 0, stream>>>(x, Xh, nx);
  cvt_wqkv_kernel<<<(1536 * 512 + 255) / 256, 256, 0, stream>>>(w_q, w_kv, Wt);
  cvt_wout_kernel<<<(4096 * 4096 + 255) / 256, 256, 0, stream>>>(w_out, Wot);

  proj_kernel<<<dim3(64, 12), 256, 0, stream>>>(Xh, Wt, Qb, Kb, Vt);
  attn_kernel<<<512, 256, 0, stream>>>(Qb, Kb, Vt, AO);
  outproj_kernel<<<dim3(8, 32), 256, 0, stream>>>(AO, Wot, b_out, y);
}
